// LSTMModel_12936441496157
// MI455X (gfx1250) — 8-bit matrix operand order repaired and checked by reading, compiled, not run on hardware
//
#include <hip/hip_runtime.h>

typedef __attribute__((ext_vector_type(8)))  int   v8i;
typedef __attribute__((ext_vector_type(16))) int   v16i;
typedef __attribute__((ext_vector_type(8)))  float v8f;

#define H    50
#define TT   256
#define GP   256
#define HP   64
#define WSCALE 8.0f
#define HSCALE 4.0f
#define INVSC  (1.0f/32.0f)

__device__ inline unsigned int f32_to_fp8_manual(float x) {
  unsigned int u; __builtin_memcpy(&u, &x, 4);
  unsigned int s = (u >> 24) & 0x80u;
  unsigned int au = u & 0x7FFFFFFFu;
  float ax; __builtin_memcpy(&ax, &au, 4);
  if (ax >= 448.0f) return s | 0x7Eu;
  if (ax < 0.0009765625f) return s;
  if (ax < 0.015625f) {
    int q = (int)(ax * 512.0f + 0.5f);
    if (q > 7) q = 7;
    return s | (unsigned int)q;
  }
  unsigned int m = au + 0x00080000u;
  int e = (int)((m >> 23) & 0xFFu) - 127 + 7;
  unsigned int m3 = (m >> 20) & 0x7u;
  if (e >= 16) return s | 0x7Eu;
  if (e <= 0)  return s;
  return s | ((unsigned int)e << 3) | m3;
}

__device__ inline unsigned int f32_to_fp8(float x) {
#if __has_builtin(__builtin_amdgcn_cvt_pk_fp8_f32)
  return (unsigned int)__builtin_amdgcn_cvt_pk_fp8_f32(x, 0.0f, 0, false) & 0xFFu;
#else
  return f32_to_fp8_manual(x);
#endif
}

__device__ inline float rcp_fast(float x) {
#if __has_builtin(__builtin_amdgcn_rcpf)
  return __builtin_amdgcn_rcpf(x);
#else
  return 1.0f / x;
#endif
}
__device__ inline float sigf(float x)      { return rcp_fast(1.0f + __expf(-x)); }
__device__ inline float tanh_fast(float x) { return 2.0f * sigf(2.0f * x) - 1.0f; }

__device__ inline v8i gather_a(const unsigned char* hb, int lane) {
  int m  = lane & 15;
  int kh = (lane < 16) ? 0 : 8;
  const unsigned char* row = hb + m * HP;
  v8i a;
#pragma unroll
  for (int d = 0; d < 8; ++d) {
    int k = ((d >> 1) << 4) + ((d & 1) << 2) + kh;
    int w; __builtin_memcpy(&w, row + k, 4);
    a[d] = w;
  }
  return a;
}

__device__ inline v8i gather_b64(const unsigned char* Wp, int n0, int lane) {
  int col = n0 + (lane & 15);
  int kh  = (lane < 16) ? 0 : 8;
  v8i b;
#pragma unroll
  for (int d = 0; d < 8; ++d) {
    int kb = ((d >> 1) << 4) + kh + ((d & 1) << 2);
    unsigned int w = 0;
#pragma unroll
    for (int by = 0; by < 4; ++by)
      w |= (unsigned int)Wp[(kb + by) * GP + col] << (8 * by);
    b[d] = (int)w;
  }
  return b;
}

__device__ inline void stage_weights(unsigned char* Wp, const float* Wg, int tid) {
  for (int idx = tid; idx < HP * GP; idx += 128) {
    int k = idx >> 8;
    int n = idx & 255;
    int q = n >> 6, j = n & 63;
    float v = 0.0f;
    if (j < H && k < H) v = Wg[(q * H + j) * H + k] * WSCALE;
    Wp[k * GP + n] = (unsigned char)f32_to_fp8(v);
  }
}

__global__ __launch_bounds__(128)
void lstm2_fused_fp8(const float* __restrict__ x,
                     const float* __restrict__ Wih1, const float* __restrict__ Whh1,
                     const float* __restrict__ bih1, const float* __restrict__ bhh1,
                     const float* __restrict__ Wih2, const float* __restrict__ Whh2,
                     const float* __restrict__ bih2, const float* __restrict__ bhh2,
                     const float* __restrict__ W1,  const float* __restrict__ b1,
                     const float* __restrict__ W2,  const float* __restrict__ b2,
                     float* __restrict__ out)
{
  __shared__ union Smem {
    unsigned char wstage[HP * GP];
    struct Run {
      float x[TT * 16];
      float h2f[16 * HP];
      float part[128];
    } run;
  } U;
  __shared__ unsigned char h1b[16 * HP];
  __shared__ unsigned char h2b[16 * HP];

  const int tid  = threadIdx.x;
  const int lane = tid & 31;
  const int wave = tid >> 5;
  const int m0   = blockIdx.x * 16;

  for (int i = tid; i < 16 * HP; i += 128) { h1b[i] = 0; h2b[i] = 0; }

  v8i  bhh1f[4];
  v16i bw2f[4];
  {
    stage_weights(U.wstage, Whh1, tid); __syncthreads();
#pragma unroll
    for (int q = 0; q < 4; ++q)
      bhh1f[q] = gather_b64(U.wstage, q * 64 + wave * 16, lane);
    __syncthreads();

    stage_weights(U.wstage, Wih2, tid); __syncthreads();
    v8i blo[4];
#pragma unroll
    for (int q = 0; q < 4; ++q)
      blo[q] = gather_b64(U.wstage, q * 64 + wave * 16, lane);
    __syncthreads();

    stage_weights(U.wstage, Whh2, tid); __syncthreads();
#pragma unroll
    for (int q = 0; q < 4; ++q) {
      v8i bhi = gather_b64(U.wstage, q * 64 + wave * 16, lane);
#pragma unroll
      for (int d = 0; d < 8; ++d) { bw2f[q][d] = blo[q][d]; bw2f[q][d + 8] = bhi[d]; }
    }
    __syncthreads();
  }

  for (int idx = tid; idx < 16 * TT; idx += 128) {
    int m = idx >> 8, t = idx & 255;
    U.run.x[t * 16 + m] = x[(size_t)(m0 + m) * TT + t];
  }

  const int  j  = wave * 16 + (lane & 15);
  const bool jv = (j < H);
  const int  mb = (lane < 16) ? 0 : 8;
  float wx[4], bs1[4], bs2[4];
#pragma unroll
  for (int q = 0; q < 4; ++q) {
    int g = q * H + j;
    wx[q]  = jv ? Wih1[g]            : 0.0f;
    bs1[q] = jv ? (bih1[g] + bhh1[g]) : 0.0f;
    bs2[q] = jv ? (bih2[g] + bhh2[g]) : 0.0f;
  }

  float c1[8], c2[8], h2last[8];
#pragma unroll
  for (int v = 0; v < 8; ++v) { c1[v] = 0.0f; c2[v] = 0.0f; h2last[v] = 0.0f; }
  const v8f czero = {};

  __syncthreads();

  for (int t = 0; t < TT; ++t) {
    v8i a1  = gather_a(h1b, lane);
    v8i ah2 = gather_a(h2b, lane);
    __syncthreads();

    v8f acc[4];
#pragma unroll
    for (int q = 0; q < 4; ++q)
      acc[q] = __builtin_amdgcn_wmma_f32_16x16x64_fp8_fp8(a1, bhh1f[q], (short)0,
                                                          czero, false, false);
    float h1v[8];
#pragma unroll
    for (int v = 0; v < 8; ++v) {
      float xv = U.run.x[t * 16 + mb + v];
      float ig = acc[0][v] * INVSC + wx[0] * xv + bs1[0];
      float fg = acc[1][v] * INVSC + wx[1] * xv + bs1[1];
      float gg = acc[2][v] * INVSC + wx[2] * xv + bs1[2];
      float og = acc[3][v] * INVSC + wx[3] * xv + bs1[3];
      float iv = sigf(ig), fv = sigf(fg), gv = tanh_fast(gg), ov = sigf(og);
      c1[v]  = fv * c1[v] + iv * gv;
      h1v[v] = ov * tanh_fast(c1[v]);
    }
    if (jv) {
#pragma unroll
      for (int v = 0; v < 8; ++v)
        h1b[(mb + v) * HP + j] = (unsigned char)f32_to_fp8(h1v[v] * HSCALE);
    }
    __syncthreads();

    v8i a1n = gather_a(h1b, lane);
    v16i a2;
#pragma unroll
    for (int d = 0; d < 8; ++d) { a2[d] = a1n[d]; a2[d + 8] = ah2[d]; }
#pragma unroll
    for (int q = 0; q < 4; ++q)
      acc[q] = __builtin_amdgcn_wmma_f32_16x16x128_fp8_fp8(a2, bw2f[q], (short)0,
                                                           czero, false, false);
#pragma unroll
    for (int v = 0; v < 8; ++v) {
      float ig = acc[0][v] * INVSC + bs2[0];
      float fg = acc[1][v] * INVSC + bs2[1];
      float gg = acc[2][v] * INVSC + bs2[2];
      float og = acc[3][v] * INVSC + bs2[3];
      float iv = sigf(ig), fv = sigf(fg), gv = tanh_fast(gg), ov = sigf(og);
      c2[v]     = fv * c2[v] + iv * gv;
      h2last[v] = ov * tanh_fast(c2[v]);
    }
    if (jv) {
#pragma unroll
      for (int v = 0; v < 8; ++v)
        h2b[(mb + v) * HP + j] = (unsigned char)f32_to_fp8(h2last[v] * HSCALE);
    }
    __syncthreads();
  }

  if (jv) {
#pragma unroll
    for (int v = 0; v < 8; ++v)
      U.run.h2f[(mb + v) * HP + j] = h2last[v];
  }
  __syncthreads();
  {
    int m = tid >> 3, s = tid & 7;
    float part = 0.0f;
    for (int u = s; u < 25; u += 8) {
      float a = b1[u];
      for (int k = 0; k < H; ++k)
        a = fmaf(U.run.h2f[m * HP + k], W1[u * H + k], a);
      part += fmaxf(a, 0.0f) * W2[u];
    }
    U.run.part[tid] = part;
  }
  __syncthreads();
  if (tid < 16) {
    float s = b2[0];
#pragma unroll
    for (int i = 0; i < 8; ++i) s += U.run.part[tid * 8 + i];
    out[m0 + tid] = s;
  }
}

extern "C" void kernel_launch(void* const* d_in, const int* in_sizes, int n_in,
                              void* d_out, int out_size, void* d_ws, size_t ws_size,
                              hipStream_t stream) {
  (void)n_in; (void)d_ws; (void)ws_size; (void)out_size;
  const float* x    = (const float*)d_in[0];
  const float* Wih1 = (const float*)d_in[1];
  const float* Whh1 = (const float*)d_in[2];
  const float* bih1 = (const float*)d_in[3];
  const float* bhh1 = (const float*)d_in[4];
  const float* Wih2 = (const float*)d_in[5];
  const float* Whh2 = (const float*)d_in[6];
  const float* bih2 = (const float*)d_in[7];
  const float* bhh2 = (const float*)d_in[8];
  const float* W1   = (const float*)d_in[9];
  const float* b1   = (const float*)d_in[10];
  const float* W2   = (const float*)d_in[11];
  const float* b2   = (const float*)d_in[12];
  float* out = (float*)d_out;

  int B = in_sizes[0] / TT;
  dim3 grid((B + 15) / 16), block(128);
  lstm2_fused_fp8<<<grid, block, 0, stream>>>(x, Wih1, Whh1, bih1, bhh1,
                                              Wih2, Whh2, bih2, bhh2,
                                              W1, b1, W2, b2, out);
}
